// PointNetFeaturePropagation_85332410237540
// MI455X (gfx1250) — matrix operand order repaired and checked by reading, not compiled, not run on hardware
//
#include <hip/hip_runtime.h>

// ---------------------------------------------------------------------------
// CDNA5 (gfx1250) wave32 WMMA types
// ---------------------------------------------------------------------------
typedef __attribute__((ext_vector_type(16))) __bf16        v16bf;
typedef __attribute__((ext_vector_type(8)))  float         v8f;
typedef __attribute__((ext_vector_type(8)))  unsigned int  v8u;

// Problem constants (from setup_inputs)
#define KB   16
#define KN   4096
#define KS   1024
#define KC2  256
#define KC1  128
#define KM   (KB * KN)      /* 65536 rows */
#define KK1  (KC2 + KC1)    /* 384 */
#define KN1  256
#define KK2  256
#define KN2  128
#define BN_EPS     1e-5f
#define INTERP_EPS 1e-8f

__device__ __forceinline__ unsigned short f2bf(float f) {
  unsigned int u = __float_as_uint(f);
  unsigned int r = u + 0x7FFFu + ((u >> 16) & 1u);   // round-to-nearest-even
  return (unsigned short)(r >> 16);
}

// ---------------------------------------------------------------------------
// K1: 3-nearest-neighbor search + inverse-distance weights
// grid = KB * (KN/256), block = 256 (one target point per thread)
// ---------------------------------------------------------------------------
__global__ __launch_bounds__(256) void k_nn3(
    const float* __restrict__ txyz, const float* __restrict__ sxyz,
    int* __restrict__ idx3, float* __restrict__ w3) {
  __shared__ float sx[KS], sy[KS], sz[KS];
  const int b    = blockIdx.x >> 4;      // 16 blocks per batch (4096/256)
  const int nblk = blockIdx.x & 15;
  const float* sb = sxyz + (size_t)b * KS * 3;
  for (int s = threadIdx.x; s < KS; s += 256) {
    sx[s] = sb[3 * s + 0];
    sy[s] = sb[3 * s + 1];
    sz[s] = sb[3 * s + 2];
  }
  __syncthreads();

  const int m = b * KN + nblk * 256 + threadIdx.x;
  const float tx = txyz[3 * m + 0], ty = txyz[3 * m + 1], tz = txyz[3 * m + 2];

  float d0 = 3.4e38f, d1 = 3.4e38f, d2 = 3.4e38f;
  int   i0 = 0, i1 = 0, i2 = 0;
  for (int s = 0; s < KS; ++s) {
    const float dx = tx - sx[s], dy = ty - sy[s], dz = tz - sz[s];
    const float d = dx * dx + dy * dy + dz * dz;
    if (d < d2) {
      if (d < d0)      { d2 = d1; i2 = i1; d1 = d0; i1 = i0; d0 = d; i0 = s; }
      else if (d < d1) { d2 = d1; i2 = i1; d1 = d;  i1 = s; }
      else             { d2 = d;  i2 = s; }
    }
  }
  const float w0 = 1.f / (d0 + INTERP_EPS);
  const float w1 = 1.f / (d1 + INTERP_EPS);
  const float w2 = 1.f / (d2 + INTERP_EPS);
  const float inv = 1.f / (w0 + w1 + w2);
  idx3[3 * m + 0] = i0; idx3[3 * m + 1] = i1; idx3[3 * m + 2] = i2;
  w3[3 * m + 0] = w0 * inv; w3[3 * m + 1] = w1 * inv; w3[3 * m + 2] = w2 * inv;
}

// ---------------------------------------------------------------------------
// K2: gather+interpolate features, concat skip, quantize to bf16 X [M, 384]
// grid = KM, block = 128
// ---------------------------------------------------------------------------
__global__ __launch_bounds__(128) void k_build_x(
    const float* __restrict__ sfeat, const float* __restrict__ skip,
    const int* __restrict__ idx3, const float* __restrict__ w3,
    unsigned short* __restrict__ X) {
  const int m = blockIdx.x;
  const int b = m >> 12;  // m / 4096
  const int   i0 = idx3[3 * m + 0], i1 = idx3[3 * m + 1], i2 = idx3[3 * m + 2];
  const float w0 = w3[3 * m + 0],  w1 = w3[3 * m + 1],  w2 = w3[3 * m + 2];
  const float* f0 = sfeat + ((size_t)b * KS + i0) * KC2;
  const float* f1 = sfeat + ((size_t)b * KS + i1) * KC2;
  const float* f2 = sfeat + ((size_t)b * KS + i2) * KC2;
  unsigned short* xr = X + (size_t)m * KK1;
  for (int c = threadIdx.x; c < KC2; c += 128)
    xr[c] = f2bf(w0 * f0[c] + w1 * f1[c] + w2 * f2[c]);
  for (int c = threadIdx.x; c < KC1; c += 128)
    xr[KC2 + c] = f2bf(skip[(size_t)m * KC1 + c]);
}

// ---------------------------------------------------------------------------
// Weight f32 -> bf16 / zero-fill helpers
// ---------------------------------------------------------------------------
__global__ void k_f2bf(const float* __restrict__ in, unsigned short* __restrict__ out, int n) {
  const int i = blockIdx.x * blockDim.x + threadIdx.x;
  if (i < n) out[i] = f2bf(in[i]);
}

__global__ void k_zero(float* __restrict__ p, int n) {
  const int i = blockIdx.x * blockDim.x + threadIdx.x;
  if (i < n) p[i] = 0.f;
}

// ---------------------------------------------------------------------------
// WMMA GEMM:  Y[M, Ndim] = X[M, KDIM](bf16) * W[Ndim, KDIM]^T(bf16), f32 accum
//
// block = 256 (8 waves). The block's 64 weight columns are staged once in
// LDS (bank-padded); each wave then owns a 32(M) x 64(N) tile: 2 A fragments
// reuse 4 LDS-resident B fragments -> 8 WMMAs per k-slab against
// 4 global b128 (A) + 8 ds_load_b128 (B).
//
// grid = (Ndim/64, M/256)
//
// Fragment layouts per CDNA5 ISA 7.12.2 (16-bit, wave32):
//   A 16x32 : lane L (L<16) = row L ; VGPR j holds K = (j>>2)*16 + (j&3)*2
//             + 8*(lane>=16)  -> two contiguous 16B loads per lane
//   B 32x16 : lane L (L<16) = col L ; VGPR j holds K = (j>>2)*16 + (j&3)*2
//             + 8*(lane>=16)  -> two 16B LDS loads per lane (same K order as A)
//   D 16x16 : lane = col + 16*(m>=8), VGPR r -> m = r + 8*(lane>=16)
// ---------------------------------------------------------------------------
template <int KDIM>
__global__ __launch_bounds__(256) void k_gemm_wmma(
    const unsigned short* __restrict__ Xb, const unsigned short* __restrict__ Wb,
    float* __restrict__ Y, int Ndim) {
  // Column stride in LDS: KDIM/2 dwords + 4 pad dwords -> consecutive columns
  // start 4 banks apart; a b128 read by 16 lanes covers disjoint bank quartets.
  constexpr int SSTRIDE = (KDIM >> 1) + 4;
  __shared__ unsigned int sB[64 * SSTRIDE];

  const int wave  = threadIdx.x >> 5;
  const int lane  = threadIdx.x & 31;
  const int l15   = lane & 15;
  const int khalf = lane >> 4;
  const int tileM = blockIdx.y * 256 + wave * 32;
  const int tileN = blockIdx.x * 64;

  // ---- stage the 64 x KDIM bf16 weight tile into LDS --------------------
  {
    constexpr int CHUNKS_PER_COL = KDIM >> 3;          // uint4 chunks per column
    constexpr int TOTAL = 64 * CHUNKS_PER_COL;
    const unsigned int* wsrc = (const unsigned int*)Wb + ((size_t)tileN * KDIM >> 1);
    for (int i = threadIdx.x; i < TOTAL; i += 256) {
      const int col = i / CHUNKS_PER_COL;
      const int off = i - col * CHUNKS_PER_COL;        // uint4 index within column
      *(uint4*)(sB + col * SSTRIDE + off * 4) =
          *(const uint4*)(wsrc + col * (KDIM >> 1) + off * 4);
    }
  }
  __syncthreads();

  const unsigned int* a0 = (const unsigned int*)Xb + ((size_t)(tileM + l15) * KDIM >> 1);
  const unsigned int* a1 = (const unsigned int*)Xb + ((size_t)(tileM + 16 + l15) * KDIM >> 1);
  const unsigned int* bls[4];
#pragma unroll
  for (int g = 0; g < 4; ++g)
    bls[g] = sB + (g * 16 + l15) * SSTRIDE;

  v8f acc0[4] = {};
  v8f acc1[4] = {};

#pragma unroll
  for (int kb = 0; kb < KDIM; kb += 32) {
    const int kd = kb >> 1;  // dword offset of this K-slab

    // A fragments (global, L2-resident): dwords 0-3 = K[kb+8*khalf .. +8),
    // dwords 4-7 = K[kb+16+8*khalf .. +8)
    const uint4 a0L = *(const uint4*)(a0 + kd + 4 * khalf);
    const uint4 a0H = *(const uint4*)(a0 + kd + 8 + 4 * khalf);
    const uint4 a1L = *(const uint4*)(a1 + kd + 4 * khalf);
    const uint4 a1H = *(const uint4*)(a1 + kd + 8 + 4 * khalf);
    union { v8u u; v16bf v; } A0, A1;
    A0.u = (v8u){a0L.x, a0L.y, a0L.z, a0L.w, a0H.x, a0H.y, a0H.z, a0H.w};
    A1.u = (v8u){a1L.x, a1L.y, a1L.z, a1L.w, a1H.x, a1H.y, a1H.z, a1H.w};

    if (kb + 32 < KDIM) {
      __builtin_prefetch((const void*)(a0 + kd + 16), 0, 3);  // global_prefetch_b8
      __builtin_prefetch((const void*)(a1 + kd + 16), 0, 3);
    }

#pragma unroll
    for (int g = 0; g < 4; ++g) {
      // B fragment from LDS: dwords 0-3 = K[kb+8*khalf .. +8), dwords 4-7 = K[kb+16+8*khalf .. +8)
      const unsigned int* p = bls[g] + kd + 4 * khalf;
      const uint4 bLo = *(const uint4*)(p);
      const uint4 bHi = *(const uint4*)(p + 8);
      union { v8u u; v16bf v; } Bv;
      Bv.u = (v8u){bLo.x, bLo.y, bLo.z, bLo.w, bHi.x, bHi.y, bHi.z, bHi.w};

      acc0[g] = __builtin_amdgcn_wmma_f32_16x16x32_bf16(
          false, A0.v, false, Bv.v, (short)0, acc0[g], false, false);
      acc1[g] = __builtin_amdgcn_wmma_f32_16x16x32_bf16(
          false, A1.v, false, Bv.v, (short)0, acc1[g], false, false);
    }
  }

#pragma unroll
  for (int g = 0; g < 4; ++g) {
    const int n = tileN + g * 16 + l15;
#pragma unroll
    for (int r = 0; r < 8; ++r) {
      Y[(size_t)(tileM + 8 * khalf + r) * Ndim + n]      = acc0[g][r];
      Y[(size_t)(tileM + 16 + 8 * khalf + r) * Ndim + n] = acc1[g][r];
    }
  }
}

// ---------------------------------------------------------------------------
// Batch-stat column reduction: grid = 256 blocks, block = NCH threads
// ---------------------------------------------------------------------------
template <int NCH>
__global__ void k_stats(const float* __restrict__ Y,
                        float* __restrict__ sum, float* __restrict__ sq) {
  const int c = threadIdx.x;
  const size_t rows = KM / 256;
  const size_t r0 = (size_t)blockIdx.x * rows;
  float s = 0.f, q = 0.f;
  for (size_t r = r0; r < r0 + rows; ++r) {
    const float v = Y[r * NCH + c];
    s += v;
    q += v * v;
  }
  atomicAdd(&sum[c], s);
  atomicAdd(&sq[c], q);
}

__global__ void k_bnfin(const float* __restrict__ sum, const float* __restrict__ sq,
                        const float* __restrict__ gamma, const float* __restrict__ beta,
                        float* __restrict__ scale, float* __restrict__ shift, int nch) {
  const int c = blockIdx.x * blockDim.x + threadIdx.x;
  if (c >= nch) return;
  const float inv_m = 1.f / (float)KM;
  const float mean = sum[c] * inv_m;
  const float var  = sq[c] * inv_m - mean * mean;
  const float sc   = gamma[c] * rsqrtf(var + BN_EPS);
  scale[c] = sc;
  shift[c] = beta[c] - mean * sc;
}

// BN + ReLU, requantize to bf16 (layer-1 output feeds GEMM2)
template <int NCH>
__global__ void k_apply_bf16(const float* __restrict__ Y, const float* __restrict__ scale,
                             const float* __restrict__ shift, unsigned short* __restrict__ Xout) {
  const size_t i = (size_t)blockIdx.x * blockDim.x + threadIdx.x;
  const int c = (int)(i & (NCH - 1));
  float v = Y[i] * scale[c] + shift[c];
  Xout[i] = f2bf(v > 0.f ? v : 0.f);
}

// BN + ReLU, f32 final output
template <int NCH>
__global__ void k_apply_f32(const float* __restrict__ Y, const float* __restrict__ scale,
                            const float* __restrict__ shift, float* __restrict__ out) {
  const size_t i = (size_t)blockIdx.x * blockDim.x + threadIdx.x;
  const int c = (int)(i & (NCH - 1));
  const float v = Y[i] * scale[c] + shift[c];
  out[i] = v > 0.f ? v : 0.f;
}

// ---------------------------------------------------------------------------
// Host-side pipeline
// ---------------------------------------------------------------------------
extern "C" void kernel_launch(void* const* d_in, const int* in_sizes, int n_in,
                              void* d_out, int out_size, void* d_ws, size_t ws_size,
                              hipStream_t stream) {
  (void)in_sizes; (void)n_in; (void)out_size; (void)ws_size;

  const float* target_xyz = (const float*)d_in[0];
  const float* source_xyz = (const float*)d_in[1];
  const float* source_fea = (const float*)d_in[2];
  const float* skip_fea   = (const float*)d_in[3];
  const float* W1     = (const float*)d_in[4];
  const float* gamma1 = (const float*)d_in[5];
  const float* beta1  = (const float*)d_in[6];
  const float* W2     = (const float*)d_in[7];
  const float* gamma2 = (const float*)d_in[8];
  const float* beta2  = (const float*)d_in[9];
  float* out = (float*)d_out;

  char* ws = (char*)d_ws;
  // Workspace layout (X region reused for X2, Y region reused for Y2):
  const size_t offX   = 0;                                 // bf16 [M,384] -> later bf16 [M,256]
  const size_t offY   = (size_t)KM * KK1 * 2;              // f32  [M,256] -> later f32  [M,128]
  const size_t offIdx = offY + (size_t)KM * KN1 * 4;
  const size_t offW3  = offIdx + (size_t)KM * 3 * 4;
  const size_t offW1b = offW3 + (size_t)KM * 3 * 4;
  const size_t offW2b = offW1b + (size_t)KN1 * KK1 * 2;
  const size_t offSt  = offW2b + (size_t)KN2 * KK2 * 2;

  unsigned short* X   = (unsigned short*)(ws + offX);
  float*          Y   = (float*)(ws + offY);
  int*            idx = (int*)(ws + offIdx);
  float*          w3  = (float*)(ws + offW3);
  unsigned short* W1b = (unsigned short*)(ws + offW1b);
  unsigned short* W2b = (unsigned short*)(ws + offW2b);
  float* st   = (float*)(ws + offSt);
  float *sum1 = st,        *sq1 = st + 256,  *scl1 = st + 512,  *sft1 = st + 768;
  float *sum2 = st + 1024, *sq2 = st + 1152, *scl2 = st + 1280, *sft2 = st + 1408;

  // 0) zero BN accumulators
  k_zero<<<(1536 + 255) / 256, 256, 0, stream>>>(st, 1536);

  // 1) 3-NN + weights
  k_nn3<<<KB * (KN / 256), 256, 0, stream>>>(target_xyz, source_xyz, idx, w3);

  // 2) weights to bf16
  k_f2bf<<<(KN1 * KK1 + 255) / 256, 256, 0, stream>>>(W1, W1b, KN1 * KK1);
  k_f2bf<<<(KN2 * KK2 + 255) / 256, 256, 0, stream>>>(W2, W2b, KN2 * KK2);

  // 3) build X = [interp(256) | skip(128)] in bf16
  k_build_x<<<KM, 128, 0, stream>>>(source_fea, skip_fea, idx, w3, X);

  // 4) GEMM1 (bf16 WMMA, f32 accum): Y1 = X * W1^T
  k_gemm_wmma<KK1><<<dim3(KN1 / 64, KM / 256), 256, 0, stream>>>(X, W1b, Y, KN1);

  // 5) BN1 stats + finalize + apply (ReLU, requantize bf16 into X region)
  k_stats<KN1><<<256, KN1, 0, stream>>>(Y, sum1, sq1);
  k_bnfin<<<1, 256, 0, stream>>>(sum1, sq1, gamma1, beta1, scl1, sft1, KN1);
  k_apply_bf16<KN1><<<(unsigned)((size_t)KM * KN1 / 256), 256, 0, stream>>>(Y, scl1, sft1, X);

  // 6) GEMM2: Y2 = X2 * W2^T (Y region reused)
  k_gemm_wmma<KK2><<<dim3(KN2 / 64, KM / 256), 256, 0, stream>>>(X, W2b, Y, KN2);

  // 7) BN2 stats + finalize + apply (ReLU, f32 -> d_out)
  k_stats<KN2><<<256, KN2, 0, stream>>>(Y, sum2, sq2);
  k_bnfin<<<1, 128, 0, stream>>>(sum2, sq2, gamma2, beta2, scl2, sft2, KN2);
  k_apply_f32<KN2><<<(unsigned)((size_t)KM * KN2 / 256), 256, 0, stream>>>(Y, scl2, sft2, out);
}
